// GRUModel_34033320853558
// MI455X (gfx1250) — compile-verified, matrix operand order checked by reading, not run on hardware
//
#include <hip/hip_runtime.h>

#define H_  50
#define HP  64
#define T_  512
#define NL  4
#define OSTRIDE 72   // padded bf16 row stride for LDS bank spread (multiple of 8)

typedef __attribute__((ext_vector_type(8)))  __bf16 v8bf;
typedef __attribute__((ext_vector_type(16))) __bf16 v16bf;
typedef __attribute__((ext_vector_type(8)))  float  v8f;

__device__ __forceinline__ float fast_sigmoid(float x) {
  return __builtin_amdgcn_rcpf(1.0f + __builtin_amdgcn_exp2f(-1.44269504f * x));
}
__device__ __forceinline__ float fast_tanh(float x) {
  return 2.0f * fast_sigmoid(2.0f * x) - 1.0f;
}

// Build a v16bf WMMA fragment for one 16x32 K-block.
// Per ISA 16-bit A/B layout: a lane holds K runs [8*half..8*half+7] and
// [16+8*half..16+8*half+7] (plus 32*kb), i.e. two contiguous 16B chunks.
__device__ __forceinline__ v16bf load_frag(const __bf16* rowbase, int half, int kb) {
  const v8bf* p0 = reinterpret_cast<const v8bf*>(rowbase + kb * 32 + half * 8);
  const v8bf* p1 = reinterpret_cast<const v8bf*>(rowbase + kb * 32 + 16 + half * 8);
  v8bf a = *p0, b = *p1;
  v16bf r;
#pragma unroll
  for (int i = 0; i < 8; ++i) { r[i] = a[i]; r[i + 8] = b[i]; }
  return r;
}

#define WMMA_BF16(A, B, C) \
  __builtin_amdgcn_wmma_f32_16x16x32_bf16(false, (A), false, (B), (short)0, (C), false, false)

// ---------------------------------------------------------------------------
// Pack kernel: pad weights 50->64 in both dims, convert to bf16, per-gate
// blocks [layer][gate][j][k] row-major k. Biases stay f32 (padded -> 0 so the
// padded hidden units stay exactly 0 through the recurrence).
// ---------------------------------------------------------------------------
__global__ void gru_pack(const float* __restrict__ Wih0, const float* __restrict__ Whh0,
                         const float* __restrict__ bih0, const float* __restrict__ bhh0,
                         const float* __restrict__ Wih,  const float* __restrict__ Whh,
                         const float* __restrict__ bih,  const float* __restrict__ bhh,
                         __bf16* __restrict__ whhP, __bf16* __restrict__ wihP,
                         float* __restrict__ bihP, float* __restrict__ bhhP,
                         float* __restrict__ w0P) {
  const int i = blockIdx.x * blockDim.x + threadIdx.x;
  if (i < NL * 3 * HP * HP) {               // recurrent weights, all 4 layers
    int k = i & 63, jj = (i >> 6) & 63, lg = i >> 12;
    int l = lg / 3, g = lg % 3;
    float v = 0.0f;
    if (jj < H_ && k < H_)
      v = (l == 0) ? Whh0[(g * H_ + jj) * H_ + k]
                   : Whh[(size_t)((l - 1) * 3 * H_ + g * H_ + jj) * H_ + k];
    whhP[i] = (__bf16)v;
  }
  if (i < 3 * 3 * HP * HP) {                // input weights, layers 1..3
    int k = i & 63, jj = (i >> 6) & 63, lg = i >> 12;
    int l1 = lg / 3, g = lg % 3;
    float v = 0.0f;
    if (jj < H_ && k < H_)
      v = Wih[(size_t)(l1 * 3 * H_ + g * H_ + jj) * H_ + k];
    wihP[i] = (__bf16)v;
  }
  if (i < NL * 3 * HP) {                    // biases (f32)
    int jj = i & 63, lg = i >> 6;
    int l = lg / 3, g = lg % 3;
    float vi = 0.0f, vh = 0.0f;
    if (jj < H_) {
      if (l == 0) { vi = bih0[g * H_ + jj]; vh = bhh0[g * H_ + jj]; }
      else        { vi = bih[(l - 1) * 3 * H_ + g * H_ + jj];
                    vh = bhh[(l - 1) * 3 * H_ + g * H_ + jj]; }
    }
    bihP[i] = vi; bhhP[i] = vh;
  }
  if (i < 3 * HP) {                         // layer-0 input weights (D=1)
    int jj = i & 63, g = i >> 6;
    w0P[i] = (jj < H_) ? Wih0[g * H_ + jj] : 0.0f;
  }
}

// ---------------------------------------------------------------------------
// Fused 4-layer GRU, time-skewed pipeline. 16 waves/WG:
//   wave = layer*4 + u ; wave owns hidden units [16u, 16u+16) of its layer.
// ---------------------------------------------------------------------------
__global__ __launch_bounds__(512) void gru_fused(
    const float* __restrict__ x,
    const __bf16* __restrict__ whhP, const __bf16* __restrict__ wihP,
    const float* __restrict__ bihP, const float* __restrict__ bhhP,
    const float* __restrict__ w0P,
    const float* __restrict__ fcw, const float* __restrict__ fcb,
    float* __restrict__ out) {
  __shared__ __align__(16) float  hS[NL][16][HP];            // f32 hidden state
  __shared__ __align__(16) __bf16 oS[2][NL][16][OSTRIDE];    // ping-pong bf16 outputs

  const int tid   = threadIdx.x;
  const int wave  = tid >> 5;
  const int lane  = tid & 31;
  const int layer = wave >> 2;
  const int u     = wave & 3;
  const int half  = lane >> 4;
  const int n16   = lane & 15;
  const int j     = u * 16 + n16;          // hidden unit owned by this lane
  const int wgRow = blockIdx.x * 16;

  for (int i = tid; i < NL * 16 * HP; i += blockDim.x) ((float*)hS)[i] = 0.0f;
  for (int i = tid; i < 2 * NL * 16 * OSTRIDE; i += blockDim.x) ((__bf16*)oS)[i] = (__bf16)0.0f;

  // ---- preload B-fragments (time-invariant) into registers ----
  v16bf Bh[3][2], Bx[3][2];
#pragma unroll
  for (int g = 0; g < 3; ++g) {
    const __bf16* rh = whhP + (size_t)((layer * 3 + g) * HP + j) * HP;
#pragma unroll
    for (int kb = 0; kb < 2; ++kb) Bh[g][kb] = load_frag(rh, half, kb);
  }
  if (layer > 0) {
#pragma unroll
    for (int g = 0; g < 3; ++g) {
      const __bf16* rx = wihP + (size_t)(((layer - 1) * 3 + g) * HP + j) * HP;
#pragma unroll
      for (int kb = 0; kb < 2; ++kb) Bx[g][kb] = load_frag(rx, half, kb);
    }
  } else {
#pragma unroll
    for (int g = 0; g < 3; ++g)
#pragma unroll
      for (int kb = 0; kb < 2; ++kb) Bx[g][kb] = Bh[g][kb];   // dead values
  }

  const float biR = bihP[(layer * 3 + 0) * HP + j];
  const float biZ = bihP[(layer * 3 + 1) * HP + j];
  const float biN = bihP[(layer * 3 + 2) * HP + j];
  const float bhR = bhhP[(layer * 3 + 0) * HP + j];
  const float bhZ = bhhP[(layer * 3 + 1) * HP + j];
  const float bhN = bhhP[(layer * 3 + 2) * HP + j];

  float w0r = 0.f, w0z = 0.f, w0n = 0.f;
  float xcur[8];
  if (layer == 0) {
    w0r = w0P[0 * HP + j]; w0z = w0P[1 * HP + j]; w0n = w0P[2 * HP + j];
#pragma unroll
    for (int v = 0; v < 8; ++v) xcur[v] = x[(size_t)(wgRow + 8 * half + v) * T_];
  }

  __syncthreads();

  // ---- time-skewed pipeline: layer l handles t = s - l ----
  for (int s = 0; s < T_ + NL - 1; ++s) {
    const int rd = s & 1, wr = rd ^ 1;
    const int t = s - layer;
    if (t >= 0 && t < T_) {
      // gh = h(t-1) @ Whh^T : A from own previous output (bf16, LDS)
      v16bf Ah0 = load_frag(&oS[rd][layer][n16][0], half, 0);
      v16bf Ah1 = load_frag(&oS[rd][layer][n16][0], half, 1);
      v8f ghR = {}, ghZ = {}, ghN = {};
      ghR = WMMA_BF16(Ah0, Bh[0][0], ghR); ghR = WMMA_BF16(Ah1, Bh[0][1], ghR);
      ghZ = WMMA_BF16(Ah0, Bh[1][0], ghZ); ghZ = WMMA_BF16(Ah1, Bh[1][1], ghZ);
      ghN = WMMA_BF16(Ah0, Bh[2][0], ghN); ghN = WMMA_BF16(Ah1, Bh[2][1], ghN);

      // gx = x_in(t) @ Wih^T : layer0 is D=1 (scalar madd), else WMMA on
      // previous layer's output produced at step s-1.
      v8f gxR, gxZ, gxN;
      if (layer == 0) {
#pragma unroll
        for (int v = 0; v < 8; ++v) {
          gxR[v] = xcur[v] * w0r; gxZ[v] = xcur[v] * w0z; gxN[v] = xcur[v] * w0n;
        }
      } else {
        v16bf Ax0 = load_frag(&oS[rd][layer - 1][n16][0], half, 0);
        v16bf Ax1 = load_frag(&oS[rd][layer - 1][n16][0], half, 1);
        v8f a = {}, b = {}, c = {};
        a = WMMA_BF16(Ax0, Bx[0][0], a); a = WMMA_BF16(Ax1, Bx[0][1], a);
        b = WMMA_BF16(Ax0, Bx[1][0], b); b = WMMA_BF16(Ax1, Bx[1][1], b);
        c = WMMA_BF16(Ax0, Bx[2][0], c); c = WMMA_BF16(Ax1, Bx[2][1], c);
        gxR = a; gxZ = b; gxN = c;
      }

      // gates: all three gate values for unit j live in this lane's fragment
#pragma unroll
      for (int v = 0; v < 8; ++v) {
        const int m = 8 * half + v;                 // C layout: row = v + 8*half
        float r  = fast_sigmoid(gxR[v] + biR + ghR[v] + bhR);
        float z  = fast_sigmoid(gxZ[v] + biZ + ghZ[v] + bhZ);
        float n  = fast_tanh(gxN[v] + biN + r * (ghN[v] + bhN));
        float ho = hS[layer][m][j];
        float hn = (1.0f - z) * n + z * ho;
        hS[layer][m][j] = hn;
        oS[wr][layer][m][j] = (__bf16)hn;
      }

      if (layer == 0 && t + 1 < T_) {               // prefetch next x
#pragma unroll
        for (int v = 0; v < 8; ++v)
          xcur[v] = x[(size_t)(wgRow + 8 * half + v) * T_ + (t + 1)];
      }
    }
    __syncthreads();
  }

  // ---- FC head: sigmoid(h3(T-1) . fcw + fcb) ----
  if (wave == 0 && lane < 16) {
    float acc = fcb[0];
#pragma unroll 10
    for (int k = 0; k < H_; ++k) acc += hS[3][lane][k] * fcw[k];
    out[wgRow + lane] = fast_sigmoid(acc);
  }
}

extern "C" void kernel_launch(void* const* d_in, const int* in_sizes, int n_in,
                              void* d_out, int out_size, void* d_ws, size_t ws_size,
                              hipStream_t stream) {
  (void)n_in; (void)out_size; (void)ws_size;
  const float* x    = (const float*)d_in[0];
  const float* Wih0 = (const float*)d_in[1];
  const float* Whh0 = (const float*)d_in[2];
  const float* bih0 = (const float*)d_in[3];
  const float* bhh0 = (const float*)d_in[4];
  const float* Wih  = (const float*)d_in[5];
  const float* Whh  = (const float*)d_in[6];
  const float* bih  = (const float*)d_in[7];
  const float* bhh  = (const float*)d_in[8];
  const float* fcw  = (const float*)d_in[9];
  const float* fcb  = (const float*)d_in[10];
  float* out = (float*)d_out;

  char* ws = (char*)d_ws;
  __bf16* whhP = (__bf16*)(ws);                        // 4*3*64*64*2  = 98304 B
  __bf16* wihP = (__bf16*)(ws + 98304);                // 3*3*64*64*2  = 73728 B
  float*  bihP = (float*)(ws + 98304 + 73728);         // 4*3*64*4     =  3072 B
  float*  bhhP = (float*)(ws + 98304 + 73728 + 3072);  //              =  3072 B
  float*  w0P  = (float*)(ws + 98304 + 73728 + 6144);  // 3*64*4       =   768 B

  const int B = in_sizes[0] / T_;                      // D = 1
  const int npack = NL * 3 * HP * HP;
  gru_pack<<<(npack + 255) / 256, 256, 0, stream>>>(Wih0, Whh0, bih0, bhh0,
                                                    Wih, Whh, bih, bhh,
                                                    whhP, wihP, bihP, bhhP, w0P);
  gru_fused<<<B / 16, 512, 0, stream>>>(x, whhP, wihP, bihP, bhhP, w0P, fcw, fcb, out);
}
